// MyEnsemble_54511724921000
// MI455X (gfx1250) — matrix operand order repaired and checked by reading, not compiled, not run on hardware
//
#include <hip/hip_runtime.h>
#include <hip/hip_bf16.h>
#include <stddef.h>

// ---------------------------------------------------------------------------
// Fused MoE-ensemble head for MI455X (gfx1250, wave32, WMMA).
//
// Reference collapses algebraically to:  out = relu(x @ Wfull + bfull)
// with Wfull = W_comb @ Wc (512x100), bfull = b_comb @ Wc + bc, where W_comb
// folds the generalist head + specialist dustbin/scatter (all linear in x).
//
// Main GEMM [131072x512]x[512x100] is HBM-bandwidth bound (~321 MB traffic,
// 13.8us floor @ 23.3 TB/s; 13.4 GFLOP needs ~1 PFLOPS). We use
// v_wmma_f32_16x16x32_bf16 with a bf16 hi/lo split (xh*Wh + xh*Wl + xl*Wh,
// residual ~2^-18) to retain ~fp32 accuracy far above the bandwidth roofline.
// WMMAs are issued in 3 passes over the 7 N-tiles so dependent accumulations
// on the same C registers are 7 instructions apart (no hazard NOPs).
// ---------------------------------------------------------------------------

typedef __attribute__((ext_vector_type(16))) __bf16 v16bf;
typedef __attribute__((ext_vector_type(8)))  float  v8f;
typedef __attribute__((ext_vector_type(4)))  float  v4f;

#define D_DIM   512
#define NC      100
#define NCP     112     // 100 padded to 7 WMMA column tiles
#define S_CNT   10
#define K_CNT   10
#define KP1     11

// ---------------- prep 0: invert cluster map, fold biases -------------------
__global__ void prep_inv_bias(const int* __restrict__ clusters,
                              const float* __restrict__ bg,
                              const float* __restrict__ bs,
                              int* __restrict__ inv_s,
                              int* __restrict__ inv_k,
                              float* __restrict__ b_comb) {
    __shared__ int ss[NC], sk[NC];
    int t = threadIdx.x;
    if (t < S_CNT * K_CNT) {
        int s = t / K_CNT, k = t % K_CNT;
        int c = clusters[t];
        ss[c] = s; sk[c] = k;
    }
    __syncthreads();
    if (t < NC) {
        inv_s[t] = ss[t];
        inv_k[t] = sk[t];
        float dsum = 0.f;
        #pragma unroll
        for (int s = 0; s < S_CNT; ++s) dsum += bs[s * KP1 + K_CNT];
        int sc = ss[t], kc = sk[t];
        b_comb[t] = bg[t] + dsum * (1.f / 90.f)
                  - bs[sc * KP1 + K_CNT] * (1.f / 90.f)
                  + bs[sc * KP1 + kc];
    }
}

// ---------------- prep 1: fold generalist + specialists into W_comb ---------
__global__ void prep_wcomb(const float* __restrict__ Wg,
                           const float* __restrict__ Ws,
                           const int* __restrict__ inv_s,
                           const int* __restrict__ inv_k,
                           float* __restrict__ W_comb) {
    int idx = blockIdx.x * blockDim.x + threadIdx.x;     // D_DIM * NC
    if (idx >= D_DIM * NC) return;
    int d = idx / NC, c = idx % NC;
    float dsum = 0.f;
    #pragma unroll
    for (int s = 0; s < S_CNT; ++s) dsum += Ws[(s * D_DIM + d) * KP1 + K_CNT];
    int sc = inv_s[c], kc = inv_k[c];
    W_comb[idx] = Wg[d * NC + c]
                + dsum * (1.f / 90.f)
                - Ws[(sc * D_DIM + d) * KP1 + K_CNT] * (1.f / 90.f)
                + Ws[(sc * D_DIM + d) * KP1 + kc];
}

// ---------------- prep 2: Wfull = W_comb @ Wc, split bf16 hi/lo, bfull ------
// Whi/Wlo stored transposed as [col][k] so the GEMM's B-operand load is a
// single 32-byte contiguous read per lane (contiguous k order, same as the A load).
__global__ void prep_wfull(const float* __restrict__ W_comb,
                           const float* __restrict__ Wc,
                           const float* __restrict__ bc,
                           const float* __restrict__ b_comb,
                           __bf16* __restrict__ Whi,
                           __bf16* __restrict__ Wlo,
                           float* __restrict__ bfull) {
    int idx = blockIdx.x * blockDim.x + threadIdx.x;     // D_DIM * NCP
    if (idx >= D_DIM * NCP) return;
    int d = idx / NCP, c = idx % NCP;
    float v = 0.f;
    if (c < NC) {
        for (int j = 0; j < NC; ++j) v += W_comb[d * NC + j] * Wc[j * NC + c];
    }
    __bf16 h = (__bf16)v;
    Whi[c * D_DIM + d] = h;
    Wlo[c * D_DIM + d] = (__bf16)(v - (float)h);
    if (d == 0) {
        float bv = 0.f;
        if (c < NC) {
            bv = bc[c];
            for (int j = 0; j < NC; ++j) bv += b_comb[j] * Wc[j * NC + c];
        }
        bfull[c] = bv;
    }
}

// ---------------- main fused GEMM + bias + relu -----------------------------
// Block = 256 threads = 8 waves. Each wave: 16 rows x 112 cols (7 N-tiles).
// K loop: 16 steps of K=32; 21 bf16 WMMAs per step in 3 hazard-free passes.
__global__ __launch_bounds__(256)
void fused_gemm_relu(const float* __restrict__ x,
                     const __bf16* __restrict__ Whi,   // [NCP][D_DIM]
                     const __bf16* __restrict__ Wlo,   // [NCP][D_DIM]
                     const float* __restrict__ bfull,  // [NCP]
                     float* __restrict__ out) {        // [B][NC]
    const int lane = threadIdx.x & 31;
    const int wave = threadIdx.x >> 5;
    const int hf   = lane >> 4;          // lane half: 0 or 1
    const int l15  = lane & 15;
    const int rowBase = blockIdx.x * 128 + wave * 16;

    v8f acc[7] = {};

    // A operand: lane half 0 covers K {k0..k0+15} of row l15;
    // lane half 1 covers K {k0+16..k0+31}  (contiguous, matches B).
    const float* xrow = x + (size_t)(rowBase + l15) * D_DIM;

    for (int k0 = 0; k0 < D_DIM; k0 += 32) {
        // ---- load A (streaming, read-once -> non-temporal) ----
        const v4f* p0 = (const v4f*)(xrow + k0 + hf * 16);        // 32B aligned
        v4f a0 = __builtin_nontemporal_load(p0);
        v4f a1 = __builtin_nontemporal_load(p0 + 1);
        v4f a2 = __builtin_nontemporal_load(p0 + 2);
        v4f a3 = __builtin_nontemporal_load(p0 + 3);
        float af[16] = { a0[0], a0[1], a0[2], a0[3],  a1[0], a1[1], a1[2], a1[3],
                         a2[0], a2[1], a2[2], a2[3],  a3[0], a3[1], a3[2], a3[3] };

        // ---- split A into bf16 hi/lo ----
        v16bf ahi, alo;
        #pragma unroll
        for (int i = 0; i < 16; ++i) {
            __bf16 h = (__bf16)af[i];
            ahi[i] = h;
            alo[i] = (__bf16)(af[i] - (float)h);
        }

        // ---- load B tiles (hot in WGP$/L2, 229 KB total) ----
        v16bf bhi[7], blo[7];
        #pragma unroll
        for (int t = 0; t < 7; ++t) {
            const size_t boff = (size_t)(t * 16 + l15) * D_DIM + k0 + hf * 16;
            bhi[t] = *(const v16bf*)(Whi + boff);
            blo[t] = *(const v16bf*)(Wlo + boff);
        }

        // ---- 3 WMMA passes: same-acc reuses are 7 instructions apart ----
        #pragma unroll
        for (int t = 0; t < 7; ++t)
            acc[t] = __builtin_amdgcn_wmma_f32_16x16x32_bf16(
                         false, ahi, false, bhi[t], (short)0, acc[t], false, false);
        #pragma unroll
        for (int t = 0; t < 7; ++t)
            acc[t] = __builtin_amdgcn_wmma_f32_16x16x32_bf16(
                         false, ahi, false, blo[t], (short)0, acc[t], false, false);
        #pragma unroll
        for (int t = 0; t < 7; ++t)
            acc[t] = __builtin_amdgcn_wmma_f32_16x16x32_bf16(
                         false, alo, false, bhi[t], (short)0, acc[t], false, false);
    }

    // C layout: VGPR j, lanes 0-15 -> (row rowBase+j,   col l15)
    //                   lanes 16-31 -> (row rowBase+8+j, col l15)
    #pragma unroll
    for (int t = 0; t < 7; ++t) {
        int col = t * 16 + l15;
        if (col < NC) {
            float bias = bfull[col];
            #pragma unroll
            for (int j = 0; j < 8; ++j) {
                int row = rowBase + j + hf * 8;
                float v = acc[t][j] + bias;
                v = v > 0.f ? v : 0.f;
                __builtin_nontemporal_store(v, out + (size_t)row * NC + col);
            }
        }
    }
}

// ---------------------------------------------------------------------------
extern "C" void kernel_launch(void* const* d_in, const int* in_sizes, int n_in,
                              void* d_out, int out_size, void* d_ws, size_t ws_size,
                              hipStream_t stream) {
    const float* x        = (const float*)d_in[0];
    const float* Wg       = (const float*)d_in[1];
    const float* bg       = (const float*)d_in[2];
    const float* Ws       = (const float*)d_in[3];
    const float* bs       = (const float*)d_in[4];
    const float* Wc       = (const float*)d_in[5];
    const float* bc       = (const float*)d_in[6];
    const int*   clusters = (const int*)d_in[7];
    float* out = (float*)d_out;

    char* ws = (char*)d_ws;
    int*    inv_s  = (int*)(ws + 0);
    int*    inv_k  = (int*)(ws + 512);
    float*  b_comb = (float*)(ws + 1024);
    float*  bfull  = (float*)(ws + 2048);
    float*  W_comb = (float*)(ws + 4096);                 // 512*100*4 = 204800 B
    __bf16* Whi    = (__bf16*)(ws + 4096 + 204800);       // 112*512*2 = 114688 B
    __bf16* Wlo    = (__bf16*)(ws + 4096 + 204800 + 114688);

    const int B = in_sizes[0] / D_DIM;                    // 131072

    prep_inv_bias<<<1, 128, 0, stream>>>(clusters, bg, bs, inv_s, inv_k, b_comb);
    prep_wcomb<<<(D_DIM * NC + 255) / 256, 256, 0, stream>>>(Wg, Ws, inv_s, inv_k, W_comb);
    prep_wfull<<<(D_DIM * NCP + 255) / 256, 256, 0, stream>>>(W_comb, Wc, bc, b_comb,
                                                              Whi, Wlo, bfull);
    fused_gemm_relu<<<B / 128, 256, 0, stream>>>(x, Whi, Wlo, bfull, out);
}
